// GptOssMoE_54219667145008
// MI455X (gfx1250) — matrix operand order repaired and checked by reading, not compiled, not run on hardware
//
#include <hip/hip_runtime.h>
#include <hip/hip_bf16.h>

// ---------------- problem constants (from reference) ----------------
#define N_TOK   2048            // BS*SLEN
#define DIM     2048
#define HID     1024
#define NEXP    8
#define TOPK    2
#define NKSLOT  (N_TOK*TOPK)    // 4096 routed slots
#define BM      128             // GEMM M tile
#define BN1     64              // GEMM1 N tile
#define BN2     128             // GEMM2 N tile
#define MCAP    (NKSLOT + NEXP*BM)       // 5120: per-expert segments padded to 128
#define MTILES_CAP (MCAP/BM)             // 40 M-tiles, static grid
#define A_TILE_BYTES  (BM*32*2)          // 8192 B
#define B1_TILE_BYTES (BN1*32*2)         // 4096 B
#define B2_TILE_BYTES (BN2*32*2)         // 8192 B

typedef __bf16 bf16;
typedef __attribute__((ext_vector_type(16))) __bf16 v16bf;
typedef __attribute__((ext_vector_type(8)))  __bf16 v8bf;
typedef __attribute__((ext_vector_type(8)))  float  v8f;

__device__ __forceinline__ unsigned short f2bf(float f) {
  unsigned int u = __builtin_bit_cast(unsigned int, f);
  u += 0x7fffu + ((u >> 16) & 1u);          // round-to-nearest-even
  return (unsigned short)(u >> 16);
}

// ---- CDNA5 async global->LDS copy (ASYNCcnt-tracked DMA, no VGPR round-trip) ----
__device__ __forceinline__ void async_ld_b128(unsigned lds_off, const void* gaddr) {
  asm volatile("global_load_async_to_lds_b128 %0, %1, off"
               :: "v"(lds_off), "v"(gaddr) : "memory");
}
__device__ __forceinline__ void wait_async0() {
  asm volatile("s_wait_asynccnt 0x0" ::: "memory");
}
__device__ __forceinline__ void wait_ds0() {
  asm volatile("s_wait_dscnt 0x0" ::: "memory");
}
__device__ __forceinline__ unsigned lds_off_of(const void* p) {
  return (unsigned)(unsigned long long)p;   // LDS aperture: low 32 bits = LDS byte addr
}

// ---------------- router: scores = sigmoid(x @ gate_w^T), stable top-2 ----------------
__global__ __launch_bounds__(256)
void moe_router_kernel(const float* __restrict__ x, const float* __restrict__ gate_w,
                       int* __restrict__ sel, float* __restrict__ scr) {
  const int t = blockIdx.x, tid = threadIdx.x;
  float s[NEXP];
#pragma unroll
  for (int e = 0; e < NEXP; ++e) s[e] = 0.f;
  for (int d = tid; d < DIM; d += 256) {
    const float xv = x[(long)t*DIM + d];
#pragma unroll
    for (int e = 0; e < NEXP; ++e) s[e] += xv * gate_w[e*DIM + d];
  }
#pragma unroll
  for (int e = 0; e < NEXP; ++e)
    for (int off = 16; off > 0; off >>= 1) s[e] += __shfl_xor(s[e], off, 32);
  __shared__ float part[8][NEXP];
  __shared__ float sc[NEXP];
  const int w = tid >> 5, lane = tid & 31;
  if (lane == 0)
    for (int e = 0; e < NEXP; ++e) part[w][e] = s[e];
  __syncthreads();
  if (tid < NEXP) {
    float tot = 0.f;
    for (int ww = 0; ww < 8; ++ww) tot += part[ww][tid];
    sc[tid] = 1.f / (1.f + __expf(-tot));
  }
  __syncthreads();
  if (tid == 0) {
    float v[NEXP];
#pragma unroll
    for (int e = 0; e < NEXP; ++e) v[e] = sc[e];
    for (int j = 0; j < TOPK; ++j) {
      int bi = 0; float bv = v[0];
      for (int e = 1; e < NEXP; ++e) if (v[e] > bv) { bv = v[e]; bi = e; }  // ties: lowest idx
      sel[t*TOPK + j] = bi; scr[t*TOPK + j] = bv;
      v[bi] = -1.f;
    }
  }
}

// ---------------- stable bucket-by-expert with 128-row tile padding ----------------
__global__ void moe_order_kernel(const int* __restrict__ sel, const float* __restrict__ scr,
                                 int* __restrict__ row_token, float* __restrict__ row_score,
                                 int* __restrict__ token_row, int* __restrict__ tile_expert) {
  if (threadIdx.x != 0 || blockIdx.x != 0) return;
  int counts[NEXP], fill[NEXP];
  for (int e = 0; e < NEXP; ++e) counts[e] = 0;
  for (int s = 0; s < NKSLOT; ++s) counts[sel[s]]++;
  int base = 0;
  for (int e = 0; e < NEXP; ++e) {
    fill[e] = base;
    const int nt = (counts[e] + BM - 1) / BM;
    for (int ti = 0; ti < nt; ++ti) tile_expert[base/BM + ti] = e;
    base += nt * BM;
  }
  for (int ti = base/BM; ti < MTILES_CAP; ++ti) tile_expert[ti] = -1;
  for (int r = 0; r < MCAP; ++r) { row_token[r] = -1; row_score[r] = 0.f; }
  for (int s = 0; s < NKSLOT; ++s) {            // monotonic s -> stable order
    const int e = sel[s], r = fill[e]++;
    row_token[r] = s / TOPK;
    row_score[r] = scr[s];
    token_row[s] = r;
  }
}

// ---------------- fp32 -> bf16 cast (weights / activations) ----------------
__global__ void cast_f32_bf16_kernel(const float* __restrict__ src, bf16* __restrict__ dst, long n) {
  for (long i = blockIdx.x*(long)blockDim.x + threadIdx.x; i < n; i += (long)gridDim.x*blockDim.x)
    dst[i] = __builtin_bit_cast(bf16, f2bf(src[i]));
}

// routed_x[r,:] = x[token(r),:] * score(r)  (pad rows -> 0), bf16
__global__ void build_routed_kernel(const float* __restrict__ x, const int* __restrict__ row_token,
                                    const float* __restrict__ row_score, bf16* __restrict__ rx) {
  const long total = (long)MCAP * DIM;
  for (long i = blockIdx.x*(long)blockDim.x + threadIdx.x; i < total; i += (long)gridDim.x*blockDim.x) {
    const int r = (int)(i / DIM), d = (int)(i % DIM);
    const int t = row_token[r];
    const float v = (t >= 0) ? x[(long)t*DIM + d] * row_score[r] : 0.f;
    rx[i] = __builtin_bit_cast(bf16, f2bf(v));
  }
}

// ---------------- WMMA helpers: fragment assembly from LDS tiles ----------------
// A (16x32 bf16): lane half h, element e -> K = h*16 + e
// B (32x16 bf16): lane half h, element e -> K = h*16 + e
__device__ __forceinline__ v16bf frag_a(const bf16* lds, int row, int half) {
  const v8bf lo = *reinterpret_cast<const v8bf*>(&lds[row*32 + half*16]);
  const v8bf hi = *reinterpret_cast<const v8bf*>(&lds[row*32 + half*16 + 8]);
  v16bf a;
#pragma unroll
  for (int i = 0; i < 8; ++i) { a[i] = lo[i]; a[i+8] = hi[i]; }
  return a;
}
__device__ __forceinline__ v16bf frag_b(const bf16* lds, int n, int half) {
  const v8bf* p = reinterpret_cast<const v8bf*>(&lds[n*32 + half*16]);
  const v8bf lo = p[0], hi = p[1];
  v16bf b;
#pragma unroll
  for (int i = 0; i < 8; ++i) { b[i] = lo[i]; b[i+8] = hi[i]; }
  return b;
}
#define WMMA_BF16(A_, B_, C_) \
  __builtin_amdgcn_wmma_f32_16x16x32_bf16(false, (A_), false, (B_), (short)0, (C_), false, false)

// ---------------- GEMM 1: H = silu(A@W1^T) * (A@W3^T), bf16 out ----------------
// block = 256 thr (8 waves, 4x2), tile 128x64, wave 32x32 per matrix,
// K-step 32, double-buffered async->LDS; 64 WMMA per barrier.
__global__ __launch_bounds__(256)
void gemm_swiglu_kernel(const bf16* __restrict__ A, const bf16* __restrict__ W1,
                        const bf16* __restrict__ W3, bf16* __restrict__ H,
                        const int* __restrict__ tile_expert) {
  const int tileM = blockIdx.x, tileN = blockIdx.y;
  int e = 0;
  if (tile_expert) { e = tile_expert[tileM]; if (e < 0) return; }
  const bf16* w1 = W1 + (long)e * HID * DIM;
  const bf16* w3 = W3 + (long)e * HID * DIM;

  __shared__ __align__(16) bf16 lA [2][BM*32];
  __shared__ __align__(16) bf16 lB1[2][BN1*32];
  __shared__ __align__(16) bf16 lB3[2][BN1*32];

  const int tid = threadIdx.x, w = tid >> 5, lane = tid & 31;
  const int wm = w >> 1, wn = w & 1;               // 4x2 waves
  const int half = lane >> 4, lr = lane & 15;
  const int aRow = tid >> 1, aK = (tid & 1) * 16;  // A: 128 rows x 2x16 bf16
  const int bRow = tid >> 2, bK = (tid & 3) * 8;   // B: 64 rows x 4x8 bf16
  const long rowM0 = (long)tileM * BM;
  const long n0 = (long)tileN * BN1;

  const unsigned aoff  = lds_off_of(&lA [0][aRow*32 + aK]);
  const unsigned b1off = lds_off_of(&lB1[0][bRow*32 + bK]);
  const unsigned b3off = lds_off_of(&lB3[0][bRow*32 + bK]);
  const bf16* gA = A  + (rowM0 + aRow)*DIM + aK;
  const bf16* g1 = w1 + (n0    + bRow)*DIM + bK;
  const bf16* g3 = w3 + (n0    + bRow)*DIM + bK;

  v8f acc1[2][2], acc3[2][2];
#pragma unroll
  for (int ms = 0; ms < 2; ++ms)
#pragma unroll
    for (int ns = 0; ns < 2; ++ns)
#pragma unroll
      for (int i = 0; i < 8; ++i) { acc1[ms][ns][i] = 0.f; acc3[ms][ns][i] = 0.f; }

  // prologue: DMA first K-slab into buffer 0
  async_ld_b128(aoff,      gA);
  async_ld_b128(aoff + 16, gA + 8);
  async_ld_b128(b1off,     g1);
  async_ld_b128(b3off,     g3);

  int p = 0;
  for (int k0 = 0; k0 < DIM; k0 += 32, p ^= 1) {
    wait_async0();        // this wave's DMA into buf p landed
    wait_ds0();           // this wave's LDS reads of buf p^1 are done
    __syncthreads();      // all waves agree -> buf p valid, buf p^1 idle
    if (k0 + 32 < DIM) {  // overlap: DMA next slab into the idle buffer
      const int q = p ^ 1;
      async_ld_b128(aoff  + q*A_TILE_BYTES,      gA + (k0 + 32));
      async_ld_b128(aoff  + q*A_TILE_BYTES + 16, gA + (k0 + 40));
      async_ld_b128(b1off + q*B1_TILE_BYTES,     g1 + (k0 + 32));
      async_ld_b128(b3off + q*B1_TILE_BYTES,     g3 + (k0 + 32));
    }
    v16bf b1[2], b3[2];
#pragma unroll
    for (int ns = 0; ns < 2; ++ns) {
      b1[ns] = frag_b(lB1[p], wn*32 + ns*16 + lr, half);
      b3[ns] = frag_b(lB3[p], wn*32 + ns*16 + lr, half);
    }
#pragma unroll
    for (int ms = 0; ms < 2; ++ms) {
      const v16bf a = frag_a(lA[p], wm*32 + ms*16 + lr, half);
#pragma unroll
      for (int ns = 0; ns < 2; ++ns) {
        acc1[ms][ns] = WMMA_BF16(a, b1[ns], acc1[ms][ns]);
        acc3[ms][ns] = WMMA_BF16(a, b3[ns], acc3[ms][ns]);
      }
    }
  }
  // epilogue: h = silu(p1) * p3 -> bf16.  C/D layout: elem r -> M = r + half*8, N = lr
#pragma unroll
  for (int ms = 0; ms < 2; ++ms)
#pragma unroll
    for (int ns = 0; ns < 2; ++ns)
#pragma unroll
      for (int r = 0; r < 8; ++r) {
        const float g = acc1[ms][ns][r];
        const float hv = (g / (1.f + __expf(-g))) * acc3[ms][ns][r];
        const long row = rowM0 + wm*32 + ms*16 + r + half*8;
        const long col = n0 + wn*32 + ns*16 + lr;
        H[row*HID + col] = __builtin_bit_cast(bf16, f2bf(hv));
      }
}

// ---------------- GEMM 2: OUT = H @ W2^T, fp32 out ----------------
// block tile 128x128, wave 32x64; 64 WMMA per barrier.
__global__ __launch_bounds__(256)
void gemm_down_kernel(const bf16* __restrict__ A, const bf16* __restrict__ W2,
                      float* __restrict__ Out, const int* __restrict__ tile_expert) {
  const int tileM = blockIdx.x, tileN = blockIdx.y;
  int e = 0;
  if (tile_expert) { e = tile_expert[tileM]; if (e < 0) return; }
  const bf16* w2 = W2 + (long)e * DIM * HID;

  __shared__ __align__(16) bf16 lA[2][BM*32];
  __shared__ __align__(16) bf16 lB[2][BN2*32];

  const int tid = threadIdx.x, w = tid >> 5, lane = tid & 31;
  const int wm = w >> 1, wn = w & 1;               // 4x2 waves; wave = 32 rows x 64 cols
  const int half = lane >> 4, lr = lane & 15;
  const int aRow = tid >> 1, aK = (tid & 1) * 16;  // 128 rows x 2x16 bf16 (A and B)
  const long rowM0 = (long)tileM * BM;
  const long n0 = (long)tileN * BN2;

  const unsigned aoff = lds_off_of(&lA[0][aRow*32 + aK]);
  const unsigned boff = lds_off_of(&lB[0][aRow*32 + aK]);
  const bf16* gA = A  + (rowM0 + aRow)*HID + aK;
  const bf16* gB = w2 + (n0    + aRow)*HID + aK;

  v8f acc[2][4];
#pragma unroll
  for (int ms = 0; ms < 2; ++ms)
#pragma unroll
    for (int ns = 0; ns < 4; ++ns)
#pragma unroll
      for (int i = 0; i < 8; ++i) acc[ms][ns][i] = 0.f;

  async_ld_b128(aoff,      gA);
  async_ld_b128(aoff + 16, gA + 8);
  async_ld_b128(boff,      gB);
  async_ld_b128(boff + 16, gB + 8);

  int p = 0;
  for (int k0 = 0; k0 < HID; k0 += 32, p ^= 1) {
    wait_async0();
    wait_ds0();
    __syncthreads();
    if (k0 + 32 < HID) {
      const int q = p ^ 1;
      async_ld_b128(aoff + q*A_TILE_BYTES,      gA + (k0 + 32));
      async_ld_b128(aoff + q*A_TILE_BYTES + 16, gA + (k0 + 40));
      async_ld_b128(boff + q*B2_TILE_BYTES,     gB + (k0 + 32));
      async_ld_b128(boff + q*B2_TILE_BYTES + 16, gB + (k0 + 40));
    }
    v16bf b[4];
#pragma unroll
    for (int ns = 0; ns < 4; ++ns)
      b[ns] = frag_b(lB[p], wn*64 + ns*16 + lr, half);
#pragma unroll
    for (int ms = 0; ms < 2; ++ms) {
      const v16bf a = frag_a(lA[p], wm*32 + ms*16 + lr, half);
#pragma unroll
      for (int ns = 0; ns < 4; ++ns)
        acc[ms][ns] = WMMA_BF16(a, b[ns], acc[ms][ns]);
    }
  }
#pragma unroll
  for (int ms = 0; ms < 2; ++ms)
#pragma unroll
    for (int ns = 0; ns < 4; ++ns)
#pragma unroll
      for (int r = 0; r < 8; ++r) {
        const long row = rowM0 + wm*32 + ms*16 + r + half*8;
        const long col = n0 + wn*64 + ns*16 + lr;
        Out[row*DIM + col] = acc[ms][ns][r];
      }
}

// ---------------- combine: out[t] += routed[row(t,0)] + routed[row(t,1)] ----------------
__global__ void combine_kernel(const float* __restrict__ routed, const int* __restrict__ token_row,
                               float* __restrict__ out) {
  const long total = (long)N_TOK * DIM;
  for (long i = blockIdx.x*(long)blockDim.x + threadIdx.x; i < total; i += (long)gridDim.x*blockDim.x) {
    const int t = (int)(i / DIM), d = (int)(i % DIM);
    const long rA = token_row[t*TOPK + 0], rB = token_row[t*TOPK + 1];
    out[i] += routed[rA*DIM + d] + routed[rB*DIM + d];
  }
}

// ---------------- host orchestration ----------------
extern "C" void kernel_launch(void* const* d_in, const int* in_sizes, int n_in,
                              void* d_out, int out_size, void* d_ws, size_t ws_size,
                              hipStream_t stream) {
  const float* x      = (const float*)d_in[0];
  const float* gate_w = (const float*)d_in[1];
  const float* w1     = (const float*)d_in[2];
  const float* w2     = (const float*)d_in[3];
  const float* w3     = (const float*)d_in[4];
  const float* sw1    = (const float*)d_in[5];
  const float* sw2    = (const float*)d_in[6];
  const float* sw3    = (const float*)d_in[7];
  float* out = (float*)d_out;

  char* ws = (char*)d_ws;
  size_t off = 0;
  auto alloc = [&](size_t bytes) -> void* {
    off = (off + 255) & ~(size_t)255;
    void* p = ws + off; off += bytes; return p;
  };
  const long WSZ = (long)NEXP * HID * DIM;     // 16.7M elems per expert weight
  bf16* w1b  = (bf16*)alloc(WSZ * 2);
  bf16* w3b  = (bf16*)alloc(WSZ * 2);
  bf16* w2b  = (bf16*)alloc(WSZ * 2);
  bf16* sw1b = (bf16*)alloc((long)HID * DIM * 2);
  bf16* sw3b = (bf16*)alloc((long)HID * DIM * 2);
  bf16* sw2b = (bf16*)alloc((long)DIM * HID * 2);
  bf16* xtb  = (bf16*)alloc((long)N_TOK * DIM * 2);
  bf16* rxb  = (bf16*)alloc((long)MCAP * DIM * 2);
  bf16* Hr   = (bf16*)alloc((long)MCAP * HID * 2);
  bf16* Hs   = (bf16*)alloc((long)N_TOK * HID * 2);
  float* rout = (float*)alloc((long)MCAP * DIM * 4);
  int*   sel       = (int*)  alloc(NKSLOT * 4);
  float* scr       = (float*)alloc(NKSLOT * 4);
  int*   row_token = (int*)  alloc(MCAP * 4);
  float* row_score = (float*)alloc(MCAP * 4);
  int*   token_row = (int*)  alloc(NKSLOT * 4);
  int*   tile_exp  = (int*)  alloc(MTILES_CAP * 4);
  (void)in_sizes; (void)n_in; (void)out_size; (void)ws_size;

  // 1) router + stable expert ordering
  moe_router_kernel<<<N_TOK, 256, 0, stream>>>(x, gate_w, sel, scr);
  moe_order_kernel<<<1, 1, 0, stream>>>(sel, scr, row_token, row_score, token_row, tile_exp);

  // 2) bf16 casts (weights mostly stay resident in 192MB L2 afterwards)
  cast_f32_bf16_kernel<<<2048, 256, 0, stream>>>(w1,  w1b,  WSZ);
  cast_f32_bf16_kernel<<<2048, 256, 0, stream>>>(w3,  w3b,  WSZ);
  cast_f32_bf16_kernel<<<2048, 256, 0, stream>>>(w2,  w2b,  WSZ);
  cast_f32_bf16_kernel<<<512,  256, 0, stream>>>(sw1, sw1b, (long)HID*DIM);
  cast_f32_bf16_kernel<<<512,  256, 0, stream>>>(sw3, sw3b, (long)HID*DIM);
  cast_f32_bf16_kernel<<<512,  256, 0, stream>>>(sw2, sw2b, (long)DIM*HID);
  cast_f32_bf16_kernel<<<1024, 256, 0, stream>>>(x,   xtb,  (long)N_TOK*DIM);
  build_routed_kernel <<<2048, 256, 0, stream>>>(x, row_token, row_score, rxb);

  // 3) routed experts: SwiGLU hidden then down-proj (per-tile expert weights)
  gemm_swiglu_kernel<<<dim3(MTILES_CAP, HID/BN1), 256, 0, stream>>>(rxb, w1b, w3b, Hr, tile_exp);
  gemm_down_kernel  <<<dim3(MTILES_CAP, DIM/BN2), 256, 0, stream>>>(Hr, w2b, rout, tile_exp);

  // 4) shared expert writes d_out directly (covers every element)
  gemm_swiglu_kernel<<<dim3(N_TOK/BM, HID/BN1), 256, 0, stream>>>(xtb, sw1b, sw3b, Hs, nullptr);
  gemm_down_kernel  <<<dim3(N_TOK/BM, DIM/BN2), 256, 0, stream>>>(Hs, sw2b, out, nullptr);

  // 5) gather-combine routed contributions (deterministic, atomic-free)
  combine_kernel<<<2048, 256, 0, stream>>>(rout, token_row, out);
}
